// AttentionMechanism_5506148073614
// MI455X (gfx1250) — compile-verified, matrix operand order checked by reading, not run on hardware
//
#include <hip/hip_runtime.h>
#include <hip/hip_bf16.h>
#include <math.h>

// ---- sizes ----
#define BATCH   128
#define REGIONS 256
#define VIS_DIM 2048
#define TXT_DIM 1024
#define HID     1024

typedef __attribute__((ext_vector_type(16))) __bf16 v16bf;
typedef __attribute__((ext_vector_type(8)))  float  v8f;

__device__ __forceinline__ unsigned short f2bf(float f) {
    unsigned u = __float_as_uint(f);
    unsigned r = u + 0x7FFFu + ((u >> 16) & 1u);   // round-to-nearest-even
    return (unsigned short)(r >> 16);
}

// ---------------------------------------------------------------------------
// Kernel 1: repack Wv (HID x VIS_DIM, f32 row-major) into bf16 B-fragments.
// Per 16x16x32 tile (kt = k/32): lane ln=n&15 selects column, h = lane>>4
// selects K-group; 16 contiguous halves per (n, h):  K(h,i) = i + 8*(h+(i>=8))
// Flat output index equals the flat enumeration index.
// ---------------------------------------------------------------------------
__global__ void pack_wv_kernel(const float* __restrict__ Wv,
                               unsigned short* __restrict__ bpack) {
    int idx = blockIdx.x * 256 + threadIdx.x;   // 64*1024*32 = 2,097,152 total
    int kt  = idx >> 15;                        // / (1024*32)
    int rem = idx & 32767;
    int n   = rem >> 5;
    int q   = rem & 31;
    int h   = q >> 4;
    int i   = q & 15;
    int K   = i + 8 * (h + (i >= 8 ? 1 : 0));
    bpack[idx] = f2bf(Wv[n * VIS_DIM + kt * 32 + K]);
}

// ---------------------------------------------------------------------------
// Kernel 2: t[b,h] = text[b,:] . Wt[h,:] + bt[h]   (tiny, f32)
// ---------------------------------------------------------------------------
__global__ void text_proj_kernel(const float* __restrict__ text,
                                 const float* __restrict__ Wt,
                                 const float* __restrict__ bt,
                                 float* __restrict__ t_ws) {
    int b = blockIdx.y;
    int h = blockIdx.x * 256 + threadIdx.x;
    const float4* x = (const float4*)(text + (size_t)b * TXT_DIM);
    const float4* w = (const float4*)(Wt   + (size_t)h * TXT_DIM);
    float acc = bt[h];
    for (int d = 0; d < TXT_DIM / 4; ++d) {
        float4 a = x[d], c = w[d];
        acc += a.x * c.x + a.y * c.y + a.z * c.z + a.w * c.w;
    }
    t_ws[(size_t)b * HID + h] = acc;
}

// ---------------------------------------------------------------------------
// Kernel 3: fused  scores[b,r] = Wa . tanh(visual[b,r,:]@Wv^T + bv + t[b]) + ba
// Block: (32-region tile rt, batch b); 256 threads = 8 waves.
// Wave w owns hid columns [128w, 128w+128) as 8 N-tiles; each wave holds TWO
// A fragments (M register blocking) so every B fragment fetched from L2 feeds
// two v_wmma_f32_16x16x32_bf16 ops.
// ---------------------------------------------------------------------------
__global__ void __launch_bounds__(256)
score_kernel(const float* __restrict__ visual,
             const unsigned short* __restrict__ bpack,
             const float* __restrict__ t_ws,
             const float* __restrict__ bv,
             const float* __restrict__ Wa,
             const float* __restrict__ ba,
             float* __restrict__ scores) {
    const int rt   = blockIdx.x;          // 0..7 region tile (32 regions)
    const int b    = blockIdx.y;          // 0..127
    const int tid  = threadIdx.x;         // 0..255
    const int lane = tid & 31;
    const int wave = tid >> 5;
    const int ln   = lane & 15;
    const int hsel = lane >> 4;

    __shared__ alignas(32) unsigned short lds_a[4][2][32][16]; // substep, m-tile
    __shared__ float lds_part[8][32];

    const float* vb = visual + ((size_t)b * REGIONS + (size_t)rt * 32) * VIS_DIM;
    const v16bf* bp = (const v16bf*)bpack;

    const v8f vzero = {0.f, 0.f, 0.f, 0.f, 0.f, 0.f, 0.f, 0.f};
    v8f acc[2][8];
#pragma unroll
    for (int mt = 0; mt < 2; ++mt)
#pragma unroll
        for (int nt = 0; nt < 8; ++nt) acc[mt][nt] = vzero;

    for (int k0 = 0; k0 < VIS_DIM; k0 += 128) {
        // ---- cooperative stage: 32 regions x 128 k, f32 -> bf16 A-fragments
#pragma unroll
        for (int t = 0; t < 16; ++t) {
            int li = tid + t * 256;          // 0..4095
            int m  = li >> 7;                // region row 0..31
            int kk = li & 127;               // k within chunk
            float f = vb[(size_t)m * VIS_DIM + k0 + kk];
            int mt = m >> 4;
            int mL = m & 15;
            int s  = kk >> 5;                // K-substep 0..3
            int k2 = kk & 31;
            int g  = k2 >> 3;                // K-group 0..3
            int L  = mL + ((g & 1) ? 16 : 0);
            int i  = (k2 & 7) + ((g >= 2) ? 8 : 0);
            lds_a[s][mt][L][i] = f2bf(f);
        }
        __syncthreads();

        const int ktbase = k0 >> 5;
#pragma unroll
        for (int s = 0; s < 4; ++s) {
            v16bf a0 = *(const v16bf*)(&lds_a[s][0][lane][0]);
            v16bf a1 = *(const v16bf*)(&lds_a[s][1][lane][0]);
            int kt = ktbase + s;
#pragma unroll
            for (int nt = 0; nt < 8; ++nt) {
                int n = (wave * 8 + nt) * 16 + ln;
                v16bf bfrag = bp[(kt * HID + n) * 2 + hsel];
                acc[0][nt] = __builtin_amdgcn_wmma_f32_16x16x32_bf16(
                    false, a0, false, bfrag, (short)0, acc[0][nt], false, false);
                acc[1][nt] = __builtin_amdgcn_wmma_f32_16x16x32_bf16(
                    false, a1, false, bfrag, (short)0, acc[1][nt], false, false);
            }
        }
        __syncthreads();
    }

    // ---- epilogue: + bv + t, tanh, dot with Wa, reduce over hid ----
    // acc[mt][nt][j] = element (M = mt*16 + j + 8*hsel, N = (wave*8+nt)*16 + ln)
    float sacc[2][8];
#pragma unroll
    for (int mt = 0; mt < 2; ++mt)
#pragma unroll
        for (int j = 0; j < 8; ++j) sacc[mt][j] = 0.f;

#pragma unroll
    for (int nt = 0; nt < 8; ++nt) {
        int n = (wave * 8 + nt) * 16 + ln;
        float hb = bv[n] + t_ws[(size_t)b * HID + n];
        float wa = Wa[n];
#pragma unroll
        for (int mt = 0; mt < 2; ++mt)
#pragma unroll
            for (int j = 0; j < 8; ++j)
                sacc[mt][j] += tanhf(acc[mt][nt][j] + hb) * wa;
    }
    // reduce over the 16 column-lanes (xor over bits 0..3 stays within half-wave)
#pragma unroll
    for (int off = 1; off < 16; off <<= 1)
#pragma unroll
        for (int mt = 0; mt < 2; ++mt)
#pragma unroll
            for (int j = 0; j < 8; ++j)
                sacc[mt][j] += __shfl_xor(sacc[mt][j], off, 32);

    if (ln == 0) {
#pragma unroll
        for (int mt = 0; mt < 2; ++mt)
#pragma unroll
            for (int j = 0; j < 8; ++j)
                lds_part[wave][mt * 16 + hsel * 8 + j] = sacc[mt][j];
    }
    __syncthreads();
    if (tid < 32) {
        float tot = 0.f;
#pragma unroll
        for (int w = 0; w < 8; ++w) tot += lds_part[w][tid];
        scores[(size_t)b * REGIONS + rt * 32 + tid] = tot + ba[0];
    }
}

// ---------------------------------------------------------------------------
// Kernel 4: softmax over 256 regions + attended[b,d] = sum_r w[r]*visual[b,r,d]
// ---------------------------------------------------------------------------
__global__ void softmax_attend_kernel(const float* __restrict__ visual,
                                      const float* __restrict__ scores,
                                      float* __restrict__ attended,
                                      float* __restrict__ weights) {
    int b = blockIdx.x;
    int tid = threadIdx.x;                 // 256 threads = 256 regions
    __shared__ float red[256];
    __shared__ float wls[256];

    float s = scores[(size_t)b * REGIONS + tid];
    red[tid] = s; __syncthreads();
    for (int off = 128; off > 0; off >>= 1) {
        if (tid < off) red[tid] = fmaxf(red[tid], red[tid + off]);
        __syncthreads();
    }
    float mx = red[0]; __syncthreads();
    float e = __expf(s - mx);
    red[tid] = e; __syncthreads();
    for (int off = 128; off > 0; off >>= 1) {
        if (tid < off) red[tid] += red[tid + off];
        __syncthreads();
    }
    float w = e / red[0];
    weights[(size_t)b * REGIONS + tid] = w;
    wls[tid] = w; __syncthreads();

    const float* vb = visual + (size_t)b * REGIONS * VIS_DIM;
#pragma unroll
    for (int c = 0; c < VIS_DIM / 256; ++c) {
        int d = tid + c * 256;
        float acc = 0.f;
        for (int r = 0; r < REGIONS; ++r)
            acc += wls[r] * vb[(size_t)r * VIS_DIM + d];
        attended[(size_t)b * VIS_DIM + d] = acc;
    }
}

// ---------------------------------------------------------------------------
extern "C" void kernel_launch(void* const* d_in, const int* in_sizes, int n_in,
                              void* d_out, int out_size, void* d_ws, size_t ws_size,
                              hipStream_t stream) {
    const float* visual = (const float*)d_in[0];   // (128,256,2048)
    const float* text   = (const float*)d_in[1];   // (128,1024)
    const float* Wv     = (const float*)d_in[2];   // (1024,2048)
    const float* bv     = (const float*)d_in[3];   // (1024)
    const float* Wt     = (const float*)d_in[4];   // (1024,1024)
    const float* bt     = (const float*)d_in[5];   // (1024)
    const float* Wa     = (const float*)d_in[6];   // (1,1024)
    const float* ba     = (const float*)d_in[7];   // (1)

    float* out      = (float*)d_out;
    float* attended = out;                               // 128*2048
    float* weights  = out + (size_t)BATCH * VIS_DIM;     // 128*256

    float* t_ws          = (float*)d_ws;                         // 512 KB
    float* scores_ws     = t_ws + (size_t)BATCH * HID;           // 128 KB
    unsigned short* bpack = (unsigned short*)(scores_ws + (size_t)BATCH * REGIONS); // 4 MB

    // 1) pack Wv to bf16 fragment order (2,097,152 elements)
    pack_wv_kernel<<<8192, 256, 0, stream>>>(Wv, bpack);
    // 2) text projection
    text_proj_kernel<<<dim3(HID / 256, BATCH), 256, 0, stream>>>(text, Wt, bt, t_ws);
    // 3) fused WMMA GEMM + tanh + attention scores (32 regions per block)
    score_kernel<<<dim3(REGIONS / 32, BATCH), 256, 0, stream>>>(
        visual, bpack, t_ws, bv, Wa, ba, scores_ws);
    // 4) softmax + attended output
    softmax_attend_kernel<<<BATCH, 256, 0, stream>>>(visual, scores_ws, attended, weights);
}
